// DeepSeekMoELayer_34359738703
// MI455X (gfx1250) — matrix operand order repaired and checked by reading, not compiled, not run on hardware
//
#include <hip/hip_runtime.h>

// ---------------------------------------------------------------------------
// DeepSeek-style MoE layer for MI455X (gfx1250, wave32, WMMA bf16 path)
//   out      = silu(x@ws1)@ws2  +  sum_k cw_k * (silu(x@w1[e_k]) @ w2[e_k])
//   logits   = x @ gate_w
// v3: 128x128 block tiles (8 waves x 32x64); B transposed in LDS so every
// WMMA fragment is a ds_load_b128; A tiles via global_load_async_to_lds_b128
// (ASYNCcnt); NT-hint weight streaming; fast-rcp SiLU epilogues.
// ---------------------------------------------------------------------------

typedef __attribute__((ext_vector_type(16))) __bf16 v16bf;
typedef __attribute__((ext_vector_type(4)))  __bf16 v4bf;
typedef __attribute__((ext_vector_type(8)))  float  v8f;
typedef __attribute__((ext_vector_type(4)))  float  v4f;

#define H_DIM 2048
#define I_DIM 8192
#define E_NUM 32
#define K_TOP 4
#define M_DIM 1408
#define T_NUM 2048   // B*S

#define BM 128
#define BN 128
#define BK 32
#define LDA 40   // A LDS row stride (bf16): 80B, 16B-aligned, odd bank step
#define LDK 40   // B^T LDS row stride (bf16)
#define NTHREADS 256   // 8 waves

// fast SiLU: v * rcp(1+e^-v)  (v_exp_f32 + v_add + v_rcp + v_mul)
__device__ __forceinline__ float silu_f(float v) {
    return v * __builtin_amdgcn_rcpf(1.0f + __expf(-v));
}

// --- CDNA5 async global->LDS copy (bypasses VGPRs, tracked by ASYNCcnt) ----
__device__ __forceinline__ void async_load_b128(unsigned lds_addr, const void* gptr) {
    asm volatile("global_load_async_to_lds_b128 %0, %1, off"
                 :: "v"(lds_addr), "v"((unsigned long long)gptr)
                 : "memory");
}
__device__ __forceinline__ void wait_async0() {
    asm volatile("s_wait_asynccnt 0x0" ::: "memory");
}

// ---- WMMA fragment loaders (contiguous k order, same in A and B) --------------
// A (16x32): lane L row M=L&15; elements are one contiguous 16-elem run at
// k = 16*(L>>4)  -> two ds_load_b128.
__device__ __forceinline__ v16bf frag_a(const __bf16* As, int wm, int lane) {
    const int row  = wm + (lane & 15);
    const int half = lane >> 4;
    const __bf16* p = As + row * LDA + 16 * half;
    v16bf f;
#pragma unroll
    for (int i = 0; i < 8; ++i) { f[i] = p[i]; f[8 + i] = p[8 + i]; }
    return f;
}
// B (32x16): lane L column N=L&15, K run 16*(L>>4)..+15.  With B stored
// transposed in LDS (Bs[col][k]) this is one contiguous 32B run.
__device__ __forceinline__ v16bf frag_b(const __bf16* Bs, int wn, int lane) {
    const int col  = wn + (lane & 15);
    const int half = lane >> 4;
    const __bf16* p = Bs + col * LDK + 16 * half;
    v16bf f;
#pragma unroll
    for (int i = 0; i < 16; ++i) f[i] = p[i];
    return f;
}

#define WMMA_BF16(a, b, c) \
    __builtin_amdgcn_wmma_f32_16x16x32_bf16(false, (a), false, (b), (short)0, (c), false, false)

// One wave's 32x64 tile worth of WMMAs for the current K-slab.
__device__ __forceinline__ void wave_mma(const __bf16* As, const __bf16* Bs,
                                         int wm, int wn, int lane, v8f acc[2][4]) {
    const v16bf a0 = frag_a(As, wm, lane);
    const v16bf a1 = frag_a(As, wm + 16, lane);
#pragma unroll
    for (int j = 0; j < 4; ++j) {
        const v16bf b = frag_b(Bs, wn + 16 * j, lane);
        acc[0][j] = WMMA_BF16(a0, b, acc[0][j]);
        acc[1][j] = WMMA_BF16(a1, b, acc[1][j]);
    }
}

// B slab loader: 32(k) x 128(n) fp32 -> bf16, stored TRANSPOSED Bs[col][k].
// NT loads: weights are streamed exactly once, keep them out of L2's way.
__device__ __forceinline__ void load_b_slab(__bf16* Bs, const float* __restrict__ Bsrc,
                                            size_t ldb, int k0, int n0, int tid) {
#pragma unroll
    for (int it = 0; it < 4; ++it) {
        const int idx = tid + it * NTHREADS;
        const int r   = idx >> 5;          // 0..31  (k)
        const int c   = (idx & 31) * 4;    // 0..124 (n)
        const v4f v = __builtin_nontemporal_load(
            (const v4f*)&Bsrc[(size_t)(k0 + r) * ldb + n0 + c]);
        Bs[(c + 0) * LDK + r] = (__bf16)v[0];
        Bs[(c + 1) * LDK + r] = (__bf16)v[1];
        Bs[(c + 2) * LDK + r] = (__bf16)v[2];
        Bs[(c + 3) * LDK + r] = (__bf16)v[3];
    }
}

// A slab loader: 128(m) x 32(k) bf16 via async LDS fill; 2 threads/row, 32B each.
__device__ __forceinline__ void load_a_slab_async(unsigned lds_base, const __bf16* g_row,
                                                  int c /*0 or 16*/, int r) {
    const unsigned l = lds_base + (unsigned)((r * LDA + c) * 2);
    async_load_b128(l,      g_row + c);
    async_load_b128(l + 16, g_row + c + 8);
}

// ---------------------------------------------------------------------------
// Kernel 1: zero per-expert counters (workspace is not re-poisoned per replay)
__global__ void k_zero_counts(int* cnts) {
    if (threadIdx.x < E_NUM) cnts[threadIdx.x] = 0;
}

// ---------------------------------------------------------------------------
// Kernel 2: x fp32 -> bf16 (done once; reused by router experts + shared up)
__global__ void k_convert_x(const float* __restrict__ x, __bf16* __restrict__ xb,
                            int n4) {
    const int i = blockIdx.x * blockDim.x + threadIdx.x;
    if (i < n4) {
        const v4f v = ((const v4f*)x)[i];
        v4bf o;
        o[0] = (__bf16)v[0]; o[1] = (__bf16)v[1];
        o[2] = (__bf16)v[2]; o[3] = (__bf16)v[3];
        ((v4bf*)xb)[i] = o;
    }
}

// ---------------------------------------------------------------------------
// Kernel 3: router.  64 tokens/block; 4 threads per token, 8 experts each.
__global__ void k_router(const float* __restrict__ x, const float* __restrict__ gw,
                         float* __restrict__ logits_out, int* __restrict__ cnts,
                         int* __restrict__ tokIdx, int* __restrict__ tokDest,
                         float* __restrict__ tokW) {
    __shared__ float Xs[64][65];
    __shared__ float Gs[64][E_NUM];
    __shared__ float Ls[64][E_NUM];
    const int tid = threadIdx.x;
    const int t0  = blockIdx.x * 64;
    const int tl  = tid >> 2;
    const int eg  = (tid & 3) * 8;

    float acc[8] = {0.f, 0.f, 0.f, 0.f, 0.f, 0.f, 0.f, 0.f};

    for (int h0 = 0; h0 < H_DIM; h0 += 64) {
        for (int idx = tid; idx < 64 * 64; idx += 256) {
            const int r = idx >> 6, c = idx & 63;
            Xs[r][c] = x[(size_t)(t0 + r) * H_DIM + h0 + c];
        }
        for (int idx = tid; idx < 64 * E_NUM; idx += 256) {
            const int r = idx >> 5, c = idx & 31;
            Gs[r][c] = gw[(size_t)(h0 + r) * E_NUM + c];
        }
        __syncthreads();
#pragma unroll 4
        for (int h = 0; h < 64; ++h) {
            const float xv = Xs[tl][h];
#pragma unroll
            for (int j = 0; j < 8; ++j) acc[j] += xv * Gs[h][eg + j];
        }
        __syncthreads();
    }

#pragma unroll
    for (int j = 0; j < 8; ++j) {
        Ls[tl][eg + j] = acc[j];
        logits_out[(size_t)(t0 + tl) * E_NUM + eg + j] = acc[j];
    }
    __syncthreads();

    if (tid < 64) {
        const int t = t0 + tid;
        int   idx4[K_TOP];
        float val4[K_TOP];
        unsigned taken = 0;
#pragma unroll
        for (int k = 0; k < K_TOP; ++k) {
            float best = -3.0e38f;
            int   bi   = 0;
            for (int e = 0; e < E_NUM; ++e) {
                if (!((taken >> e) & 1u)) {
                    const float lv = Ls[tid][e];
                    if (lv > best) { best = lv; bi = e; }
                }
            }
            taken |= 1u << bi;
            idx4[k] = bi;
            val4[k] = best;
        }
        const float mx = val4[0];
        float ex[K_TOP], den = 0.f;
#pragma unroll
        for (int k = 0; k < K_TOP; ++k) { ex[k] = __expf(val4[k] - mx); den += ex[k]; }
        const float rden = __builtin_amdgcn_rcpf(den);
#pragma unroll
        for (int k = 0; k < K_TOP; ++k) {
            const int e    = idx4[k];
            const int slot = atomicAdd(&cnts[e], 1);
            tokIdx [e * T_NUM + slot] = t;
            tokDest[e * T_NUM + slot] = t * K_TOP + k;
            tokW   [e * T_NUM + slot] = ex[k] * rden;
        }
    }
}

// ---------------------------------------------------------------------------
// Kernel 4: shared up-proj  sact = silu(xb @ ws1)  (bf16 out)
__global__ void k_shared_up(const __bf16* __restrict__ xb,
                            const float* __restrict__ ws1,
                            __bf16* __restrict__ sact) {
    __shared__ __bf16 As[BM * LDA];
    __shared__ __bf16 Bs[BN * LDK];
    const int tid  = threadIdx.x;
    const int lane = tid & 31;
    const int wid  = tid >> 5;
    const int wm   = (wid & 3) * 32;
    const int wn   = (wid >> 2) * 64;
    const int m0   = blockIdx.x * BM;
    const int n0   = blockIdx.y * BN;
    const unsigned asA = (unsigned)(size_t)(void*)As;

    const int ar = tid >> 1;            // A row this thread fills
    const int ac = (tid & 1) * 16;      // A col offset (0 or 16)

    v8f acc[2][4] = {};
    for (int k0 = 0; k0 < H_DIM; k0 += BK) {
        load_a_slab_async(asA, xb + (size_t)(m0 + ar) * H_DIM + k0, ac, ar);
        load_b_slab(Bs, ws1, I_DIM, k0, n0, tid);
        wait_async0();
        __syncthreads();
        wave_mma(As, Bs, wm, wn, lane, acc);
        __syncthreads();
    }
    const int nlane = lane & 15;
    const int mhalf = (lane >> 4) * 8;
#pragma unroll
    for (int im = 0; im < 2; ++im)
#pragma unroll
        for (int in = 0; in < 4; ++in)
#pragma unroll
            for (int r = 0; r < 8; ++r) {
                const int m = m0 + wm + 16 * im + mhalf + r;
                const int n = n0 + wn + 16 * in + nlane;
                sact[(size_t)m * I_DIM + n] = (__bf16)silu_f(acc[im][in][r]);
            }
}

// ---------------------------------------------------------------------------
// Kernel 5: shared down-proj  out = sact @ ws2  (fp32 stores; defines out)
__global__ void k_shared_down(const __bf16* __restrict__ sact,
                              const float* __restrict__ ws2,
                              float* __restrict__ out) {
    __shared__ __bf16 As[BM * LDA];
    __shared__ __bf16 Bs[BN * LDK];
    const int tid  = threadIdx.x;
    const int lane = tid & 31;
    const int wid  = tid >> 5;
    const int wm   = (wid & 3) * 32;
    const int wn   = (wid >> 2) * 64;
    const int m0   = blockIdx.x * BM;
    const int n0   = blockIdx.y * BN;
    const unsigned asA = (unsigned)(size_t)(void*)As;

    const int ar = tid >> 1;
    const int ac = (tid & 1) * 16;

    v8f acc[2][4] = {};
    for (int k0 = 0; k0 < I_DIM; k0 += BK) {
        load_a_slab_async(asA, sact + (size_t)(m0 + ar) * I_DIM + k0, ac, ar);
        load_b_slab(Bs, ws2, H_DIM, k0, n0, tid);
        wait_async0();
        __syncthreads();
        wave_mma(As, Bs, wm, wn, lane, acc);
        __syncthreads();
    }
    const int nlane = lane & 15;
    const int mhalf = (lane >> 4) * 8;
#pragma unroll
    for (int im = 0; im < 2; ++im)
#pragma unroll
        for (int in = 0; in < 4; ++in)
#pragma unroll
            for (int r = 0; r < 8; ++r) {
                const int m = m0 + wm + 16 * im + mhalf + r;
                const int n = n0 + wn + 16 * in + nlane;
                out[(size_t)m * H_DIM + n] = acc[im][in][r];
            }
}

// ---------------------------------------------------------------------------
// Kernel 6: expert up-proj (grouped GEMM over gathered token rows)
//   eact[tokDest] = silu( xb[tokIdx] @ w1[e] )
__global__ void k_expert_up(const __bf16* __restrict__ xb,
                            const float* __restrict__ w1,
                            const int* __restrict__ cnts,
                            const int* __restrict__ tokIdx,
                            const int* __restrict__ tokDest,
                            __bf16* __restrict__ eact) {
    const int e    = blockIdx.x;
    const int n0   = blockIdx.y * BN;
    const int row0 = blockIdx.z * BM;
    const int cnt  = cnts[e];
    if (row0 >= cnt) return;

    __shared__ __bf16 As[BM * LDA];
    __shared__ __bf16 Bs[BN * LDK];
    __shared__ int rows[BM];
    __shared__ int dest[BM];

    const int tid  = threadIdx.x;
    const int lane = tid & 31;
    const int wid  = tid >> 5;
    const int wm   = (wid & 3) * 32;
    const int wn   = (wid >> 2) * 64;
    const unsigned asA = (unsigned)(size_t)(void*)As;

    if (tid < BM) {
        const int slot = row0 + tid;
        if (slot < cnt) {
            rows[tid] = tokIdx [e * T_NUM + slot];
            dest[tid] = tokDest[e * T_NUM + slot];
        } else {
            rows[tid] = 0;
            dest[tid] = -1;
        }
    }
    __syncthreads();

    const int ar = tid >> 1;
    const int ac = (tid & 1) * 16;
    const size_t arow = (size_t)rows[ar] * H_DIM;

    v8f acc[2][4] = {};
    for (int k0 = 0; k0 < H_DIM; k0 += BK) {
        load_a_slab_async(asA, xb + arow + k0, ac, ar);
        load_b_slab(Bs, w1 + (size_t)e * H_DIM * M_DIM, M_DIM, k0, n0, tid);
        wait_async0();
        __syncthreads();
        wave_mma(As, Bs, wm, wn, lane, acc);
        __syncthreads();
    }
    const int nlane = lane & 15;
    const int mhalf = (lane >> 4) * 8;
#pragma unroll
    for (int im = 0; im < 2; ++im)
#pragma unroll
        for (int in = 0; in < 4; ++in)
#pragma unroll
            for (int r = 0; r < 8; ++r) {
                const int mloc = wm + 16 * im + mhalf + r;
                const int d    = dest[mloc];
                if (d >= 0) {
                    const int n = n0 + wn + 16 * in + nlane;
                    eact[(size_t)d * M_DIM + n] = (__bf16)silu_f(acc[im][in][r]);
                }
            }
}

// ---------------------------------------------------------------------------
// Kernel 7: expert down-proj with routing-weight scale + atomic accumulate
//   out[tokIdx] += tokW * ( eact[tokDest] @ w2[e] )
__global__ void k_expert_down(const __bf16* __restrict__ eact,
                              const float* __restrict__ w2,
                              const int* __restrict__ cnts,
                              const int* __restrict__ tokIdx,
                              const int* __restrict__ tokDest,
                              const float* __restrict__ tokW,
                              float* __restrict__ out) {
    const int e    = blockIdx.x;
    const int n0   = blockIdx.y * BN;
    const int row0 = blockIdx.z * BM;
    const int cnt  = cnts[e];
    if (row0 >= cnt) return;

    __shared__ __bf16 As[BM * LDA];
    __shared__ __bf16 Bs[BN * LDK];
    __shared__ int   srcr[BM];
    __shared__ int   toks[BM];
    __shared__ float wrow[BM];

    const int tid  = threadIdx.x;
    const int lane = tid & 31;
    const int wid  = tid >> 5;
    const int wm   = (wid & 3) * 32;
    const int wn   = (wid >> 2) * 64;
    const unsigned asA = (unsigned)(size_t)(void*)As;

    if (tid < BM) {
        const int slot = row0 + tid;
        if (slot < cnt) {
            srcr[tid] = tokDest[e * T_NUM + slot];
            toks[tid] = tokIdx [e * T_NUM + slot];
            wrow[tid] = tokW   [e * T_NUM + slot];
        } else {
            srcr[tid] = 0;
            toks[tid] = -1;
            wrow[tid] = 0.f;
        }
    }
    __syncthreads();

    const int ar = tid >> 1;
    const int ac = (tid & 1) * 16;
    const size_t arow = (size_t)srcr[ar] * M_DIM;

    v8f acc[2][4] = {};
    for (int k0 = 0; k0 < M_DIM; k0 += BK) {
        load_a_slab_async(asA, eact + arow + k0, ac, ar);
        load_b_slab(Bs, w2 + (size_t)e * M_DIM * H_DIM, H_DIM, k0, n0, tid);
        wait_async0();
        __syncthreads();
        wave_mma(As, Bs, wm, wn, lane, acc);
        __syncthreads();
    }
    const int nlane = lane & 15;
    const int mhalf = (lane >> 4) * 8;
#pragma unroll
    for (int im = 0; im < 2; ++im)
#pragma unroll
        for (int in = 0; in < 4; ++in)
#pragma unroll
            for (int r = 0; r < 8; ++r) {
                const int mloc = wm + 16 * im + mhalf + r;
                const int t    = toks[mloc];
                if (t >= 0) {
                    const int n = n0 + wn + 16 * in + nlane;
                    atomicAdd(&out[(size_t)t * H_DIM + n],
                              wrow[mloc] * acc[im][in][r]);
                }
            }
}

// ---------------------------------------------------------------------------
extern "C" void kernel_launch(void* const* d_in, const int* in_sizes, int n_in,
                              void* d_out, int out_size, void* d_ws, size_t ws_size,
                              hipStream_t stream) {
    const float* x      = (const float*)d_in[0];   // [T, H]
    const float* gate_w = (const float*)d_in[1];   // [H, E]
    const float* w1     = (const float*)d_in[2];   // [E, H, M]
    const float* w2     = (const float*)d_in[3];   // [E, M, H]
    const float* ws1    = (const float*)d_in[4];   // [H, I]
    const float* ws2    = (const float*)d_in[5];   // [I, H]

    float* out        = (float*)d_out;                         // [T, H]
    float* out_logits = out + (size_t)T_NUM * H_DIM;           // [T, E]

    char*  ws  = (char*)d_ws;
    size_t off = 0;
    auto carve = [&](size_t bytes) -> void* {
        void* p = ws + off;
        off = (off + bytes + 255) & ~(size_t)255;
        return p;
    };
    __bf16* xb      = (__bf16*)carve((size_t)T_NUM * H_DIM * sizeof(__bf16));
    __bf16* sact    = (__bf16*)carve((size_t)T_NUM * I_DIM * sizeof(__bf16));
    __bf16* eact    = (__bf16*)carve((size_t)T_NUM * K_TOP * M_DIM * sizeof(__bf16));
    int*    cnts    = (int*)   carve((size_t)E_NUM * sizeof(int));
    int*    tokIdx  = (int*)   carve((size_t)E_NUM * T_NUM * sizeof(int));
    int*    tokDest = (int*)   carve((size_t)E_NUM * T_NUM * sizeof(int));
    float*  tokW    = (float*) carve((size_t)E_NUM * T_NUM * sizeof(float));
    (void)ws_size; (void)in_sizes; (void)n_in; (void)out_size;

    k_zero_counts<<<1, 32, 0, stream>>>(cnts);

    const int n4 = (T_NUM * H_DIM) / 4;
    k_convert_x<<<(n4 + 255) / 256, 256, 0, stream>>>(x, xb, n4);

    k_router<<<T_NUM / 64, 256, 0, stream>>>(x, gate_w, out_logits, cnts,
                                             tokIdx, tokDest, tokW);

    k_shared_up<<<dim3(T_NUM / BM, I_DIM / BN), NTHREADS, 0, stream>>>(xb, ws1, sact);

    k_shared_down<<<dim3(T_NUM / BM, H_DIM / BN), NTHREADS, 0, stream>>>(sact, ws2, out);

    k_expert_up<<<dim3(E_NUM, M_DIM / BN, T_NUM / BM), NTHREADS, 0, stream>>>(
        xb, w1, cnts, tokIdx, tokDest, eact);

    k_expert_down<<<dim3(E_NUM, H_DIM / BN, T_NUM / BM), NTHREADS, 0, stream>>>(
        eact, w2, cnts, tokIdx, tokDest, tokW, out);
}
